// CustomRankingLoss_19121194402179
// MI455X (gfx1250) — matrix operand order repaired and checked by reading, not compiled, not run on hardware
//
#include <hip/hip_runtime.h>
#include <hip/hip_bf16.h>
#include <math.h>

#define N_ITEMS 32768
#define DIMS    512
#define KSEL    16
#define LAMBDA  0.5f
#define NTILES  (N_ITEMS / 16)   // 2048
#define NWAVES  1024             // 128 blocks * 8 waves

typedef _Float16 h8   __attribute__((ext_vector_type(8)));
typedef _Float16 v16h __attribute__((ext_vector_type(16)));
typedef float    v8f  __attribute__((ext_vector_type(8)));

// ---------------------------------------------------------------- init
__global__ __launch_bounds__(256) void initK(float* __restrict__ msim,
                                             float* __restrict__ scal) {
    int i = blockIdx.x * 256 + threadIdx.x;
    if (i < N_ITEMS) msim[i] = 0.0f;
    if (i == 0) scal[2] = 0.0f;   // err accumulator
}

// ------------------------------------------------- softmax: max reduce
__global__ __launch_bounds__(256) void reduceMaxK(const float* __restrict__ pred,
                                                  float* __restrict__ part) {
    __shared__ float ss[256];
    int tid = threadIdx.x;
    ss[tid] = pred[blockIdx.x * 256 + tid];
    __syncthreads();
    for (int off = 128; off; off >>= 1) {
        if (tid < off) ss[tid] = fmaxf(ss[tid], ss[tid + off]);
        __syncthreads();
    }
    if (tid == 0) part[blockIdx.x] = ss[0];
}

__global__ __launch_bounds__(128) void finalMaxK(const float* __restrict__ part,
                                                 float* __restrict__ scal) {
    __shared__ float ss[128];
    int tid = threadIdx.x;
    ss[tid] = part[tid];
    __syncthreads();
    for (int off = 64; off; off >>= 1) {
        if (tid < off) ss[tid] = fmaxf(ss[tid], ss[tid + off]);
        __syncthreads();
    }
    if (tid == 0) scal[0] = ss[0];
}

// ------------------------------------------------- softmax: sum of exp
__global__ __launch_bounds__(256) void reduceSumK(const float* __restrict__ pred,
                                                  const float* __restrict__ scal,
                                                  float* __restrict__ part) {
    __shared__ float ss[256];
    int tid = threadIdx.x;
    float m = scal[0];
    ss[tid] = expf(pred[blockIdx.x * 256 + tid] - m);
    __syncthreads();
    for (int off = 128; off; off >>= 1) {
        if (tid < off) ss[tid] += ss[tid + off];
        __syncthreads();
    }
    if (tid == 0) part[blockIdx.x] = ss[0];
}

__global__ __launch_bounds__(128) void finalSumK(const float* __restrict__ part,
                                                 float* __restrict__ scal) {
    __shared__ float ss[128];
    int tid = threadIdx.x;
    ss[tid] = part[tid];
    __syncthreads();
    for (int off = 64; off; off >>= 1) {
        if (tid < off) ss[tid] += ss[tid + off];
        __syncthreads();
    }
    if (tid == 0) scal[1] = 1.0f / ss[0];
}

__global__ __launch_bounds__(256) void relK(const float* __restrict__ pred,
                                            const float* __restrict__ scal,
                                            float* __restrict__ rel) {
    int i = blockIdx.x * 256 + threadIdx.x;
    rel[i] = expf(pred[i] - scal[0]) * scal[1];
}

// --------------------------------------- L2-normalize rows, store f16
__global__ __launch_bounds__(256) void normConvertK(const float* __restrict__ E,
                                                    _Float16* __restrict__ En) {
    int lane = threadIdx.x & 31;
    int row  = blockIdx.x * 8 + (threadIdx.x >> 5);
    const float* src = E + (size_t)row * DIMS + lane * 16;
    float4 v[4];
    float ss = 0.0f;
#pragma unroll
    for (int j = 0; j < 4; ++j) {
        v[j] = *(const float4*)(src + j * 4);
        ss += v[j].x * v[j].x + v[j].y * v[j].y + v[j].z * v[j].z + v[j].w * v[j].w;
    }
#pragma unroll
    for (int off = 16; off; off >>= 1) ss += __shfl_xor(ss, off, 32);
    float inv = 1.0f / (sqrtf(ss) + 1e-12f);
    _Float16 out[16];
#pragma unroll
    for (int j = 0; j < 4; ++j) {
        out[j * 4 + 0] = (_Float16)(v[j].x * inv);
        out[j * 4 + 1] = (_Float16)(v[j].y * inv);
        out[j * 4 + 2] = (_Float16)(v[j].z * inv);
        out[j * 4 + 3] = (_Float16)(v[j].w * inv);
    }
    _Float16* dst = En + (size_t)row * DIMS + lane * 16;
    *(h8*)dst       = *(h8*)out;
    *((h8*)dst + 1) = *((h8*)out + 1);
}

// ------------------------------------------------ step 0: argmax(rel)
__global__ __launch_bounds__(256) void argmax0K(const float* __restrict__ rel,
                                                float* __restrict__ partS,
                                                int* __restrict__ partI) {
    __shared__ float ss[256];
    __shared__ int   si[256];
    int tid = threadIdx.x;
    int i = blockIdx.x * 256 + tid;
    ss[tid] = rel[i];
    si[tid] = i;
    __syncthreads();
    for (int off = 128; off; off >>= 1) {
        if (tid < off && ss[tid + off] > ss[tid]) { ss[tid] = ss[tid + off]; si[tid] = si[tid + off]; }
        __syncthreads();
    }
    if (tid == 0) { partS[blockIdx.x] = ss[0]; partI[blockIdx.x] = si[0]; }
}

// -------- phase A (t>=1): WMMA matvec vs selected row.
// Each wave owns exactly 2 of the 2048 tiles and processes them
// CONCURRENTLY (two accumulators) with A-fragment loads software-
// pipelined one K-chunk ahead, so the scheduler can overlap L2 load
// latency with the WMMAs instead of s_wait_loadcnt 0 per chunk.
__global__ __launch_bounds__(256) void phaseAK(const _Float16* __restrict__ En,
                                               const float* __restrict__ rel,
                                               float* __restrict__ msim,
                                               const int* __restrict__ seli,
                                               float* __restrict__ partS,
                                               int* __restrict__ partI,
                                               float decayt) {
    const int lane       = threadIdx.x & 31;
    const int waveInBlk  = threadIdx.x >> 5;
    const int waveGlobal = blockIdx.x * 8 + waveInBlk;
    const int half       = lane >> 4;        // 0: K 0-7/16-23, 1: K 8-15/24-31
    const int sub        = lane & 15;

    const int selIdx = seli[16];
    const _Float16* selRow = En + (size_t)selIdx * DIMS + half * 8;

    const int rowBase0 = waveGlobal * 16;                 // tile 0
    const int rowBase1 = (waveGlobal + NWAVES) * 16;      // tile 1
    const _Float16* aRow0 = En + (size_t)(rowBase0 + sub) * DIMS + half * 8;
    const _Float16* aRow1 = En + (size_t)(rowBase1 + sub) * DIMS + half * 8;

    v8f c0 = {}, c1 = {};
    v16h a0, a1;
    *(h8*)&a0       = *(const h8*)(aRow0);
    *((h8*)&a0 + 1) = *(const h8*)(aRow0 + 16);
    *(h8*)&a1       = *(const h8*)(aRow1);
    *((h8*)&a1 + 1) = *(const h8*)(aRow1 + 16);

#pragma unroll
    for (int k = 0; k < DIMS; k += 32) {
        v16h n0, n1;
        if (k + 32 < DIMS) {                 // prefetch next chunk (both tiles)
            *(h8*)&n0       = *(const h8*)(aRow0 + k + 32);
            *((h8*)&n0 + 1) = *(const h8*)(aRow0 + k + 48);
            *(h8*)&n1       = *(const h8*)(aRow1 + k + 32);
            *((h8*)&n1 + 1) = *(const h8*)(aRow1 + k + 48);
        }
        v16h b;
        *(h8*)&b       = *(const h8*)(selRow + k);
        *((h8*)&b + 1) = *(const h8*)(selRow + k + 16);
        c0 = __builtin_amdgcn_wmma_f32_16x16x32_f16(
                 false, a0, false, b, (short)0, c0, false, false);
        c1 = __builtin_amdgcn_wmma_f32_16x16x32_f16(
                 false, a1, false, b, (short)0, c1, false, false);
        if (k + 32 < DIMS) { a0 = n0; a1 = n1; }
    }

    // lanes 0-7 own rows base+0..7; lanes 16-23 own rows base+8..15
    float bestS = -__builtin_inff();
    int   bestI = 0;
    {
        float score = -__builtin_inff();
        int   row   = rowBase0 + half * 8 + sub;
        if (sub < 8) {
            float m = fmaxf(msim[row], c0[sub]);
            msim[row] = m;
            score = decayt * rel[row] - LAMBDA * m;      // rel==-inf masks
        }
        if (score > bestS) { bestS = score; bestI = row; }
    }
    {
        float score = -__builtin_inff();
        int   row   = rowBase1 + half * 8 + sub;
        if (sub < 8) {
            float m = fmaxf(msim[row], c1[sub]);
            msim[row] = m;
            score = decayt * rel[row] - LAMBDA * m;
        }
        if (score > bestS) { bestS = score; bestI = row; }
    }

#pragma unroll
    for (int off = 16; off; off >>= 1) {
        float os = __shfl_xor(bestS, off, 32);
        int   oi = __shfl_xor(bestI, off, 32);
        if (os > bestS) { bestS = os; bestI = oi; }
    }
    __shared__ float ss[8];
    __shared__ int   si[8];
    if (lane == 0) { ss[waveInBlk] = bestS; si[waveInBlk] = bestI; }
    __syncthreads();
    if (threadIdx.x == 0) {
        float bs = ss[0]; int bi = si[0];
        for (int w = 1; w < 8; ++w)
            if (ss[w] > bs) { bs = ss[w]; bi = si[w]; }
        partS[blockIdx.x] = bs;
        partI[blockIdx.x] = bi;
    }
}

// -------- phase B: finalize argmax, accumulate err, mask, record pick
__global__ __launch_bounds__(128) void phaseBK(const float* __restrict__ partS,
                                               const int* __restrict__ partI,
                                               float* __restrict__ rel,
                                               int* __restrict__ seli,
                                               float* __restrict__ scal,
                                               float decayt, int t) {
    __shared__ float ss[128];
    __shared__ int   si[128];
    int tid = threadIdx.x;
    ss[tid] = partS[tid];
    si[tid] = partI[tid];
    __syncthreads();
    for (int off = 64; off; off >>= 1) {
        if (tid < off && ss[tid + off] > ss[tid]) { ss[tid] = ss[tid + off]; si[tid] = si[tid + off]; }
        __syncthreads();
    }
    if (tid == 0) {
        int idx = si[0];
        scal[2] += decayt * rel[idx];   // err term (rel still unmasked here)
        rel[idx] = -__builtin_inff();   // mask for future steps
        seli[t]  = idx;
        seli[16] = idx;                 // "current" pick for next phase A
    }
}

// -------- final: S·S^T via single-wave WMMA, row-max sum, combine loss
__global__ __launch_bounds__(32) void finalK(const _Float16* __restrict__ En,
                                             const int* __restrict__ seli,
                                             const float* __restrict__ scal,
                                             float* __restrict__ out) {
    int lane = threadIdx.x;
    int half = lane >> 4;
    int sub  = lane & 15;
    int row  = seli[sub];
    const _Float16* aPtr = En + (size_t)row * DIMS + half * 8;   // A: M=sub
    const _Float16* bPtr = En + (size_t)row * DIMS + half * 8;   // B: N=sub (K-major)
    v8f c = {};
#pragma unroll
    for (int k = 0; k < DIMS; k += 32) {
        v16h a, b;
        *(h8*)&a       = *(const h8*)(aPtr + k);
        *((h8*)&a + 1) = *(const h8*)(aPtr + k + 16);
        *(h8*)&b       = *(const h8*)(bPtr + k);
        *((h8*)&b + 1) = *(const h8*)(bPtr + k + 16);
        c = __builtin_amdgcn_wmma_f32_16x16x32_f16(
                false, a, false, b, (short)0, c, false, false);
    }
    // row M = half*8 + r lives in VGPR r across the 16-lane group
    float total = 0.0f;
#pragma unroll
    for (int r = 0; r < 8; ++r) {
        float m = c[r];
#pragma unroll
        for (int off = 8; off; off >>= 1) m = fmaxf(m, __shfl_xor(m, off, 16));
        total += m;
    }
    total = (sub == 0) ? total : 0.0f;            // lane0: rows 0-7, lane16: rows 8-15
    total += __shfl_xor(total, 16, 32);
    if (lane == 0) out[0] = -scal[2] - LAMBDA * total;
}

// ----------------------------------------------------------------------
extern "C" void kernel_launch(void* const* d_in, const int* in_sizes, int n_in,
                              void* d_out, int out_size, void* d_ws, size_t ws_size,
                              hipStream_t stream) {
    (void)in_sizes; (void)n_in; (void)out_size; (void)ws_size;
    const float* pred = (const float*)d_in[0];
    const float* E    = (const float*)d_in[5];   // explanation_embeddings

    char* ws = (char*)d_ws;
    _Float16* En  = (_Float16*)ws;                               // 32 MB f16 (L2-resident)
    float* rel    = (float*)(ws + (size_t)N_ITEMS * DIMS * 2);   // 128 KB
    float* msim   = rel + N_ITEMS;                               // 128 KB
    float* partS  = msim + N_ITEMS;                              // 128 floats
    int*   partI  = (int*)(partS + 128);                         // 128 ints
    float* scal   = (float*)(partI + 128);                       // m, 1/denom, err
    int*   seli   = (int*)(scal + 8);                            // sel[16] + current
    float* out    = (float*)d_out;

    initK      <<<128, 256, 0, stream>>>(msim, scal);
    reduceMaxK <<<128, 256, 0, stream>>>(pred, partS);
    finalMaxK  <<<1, 128, 0, stream>>>(partS, scal);
    reduceSumK <<<128, 256, 0, stream>>>(pred, scal, partS);
    finalSumK  <<<1, 128, 0, stream>>>(partS, scal);
    relK       <<<128, 256, 0, stream>>>(pred, scal, rel);
    normConvertK<<<N_ITEMS / 8, 256, 0, stream>>>(E, En);

    argmax0K   <<<128, 256, 0, stream>>>(rel, partS, partI);
    phaseBK    <<<1, 128, 0, stream>>>(partS, partI, rel, seli, scal, 1.0f, 0);

    float d = 1.0f;
    for (int t = 1; t < KSEL; ++t) {
        d *= 0.85f;
        phaseAK<<<128, 256, 0, stream>>>(En, rel, msim, seli, partS, partI, d);
        phaseBK<<<1, 128, 0, stream>>>(partS, partI, rel, seli, scal, d, t);
    }
    finalK<<<1, 32, 0, stream>>>(En, seli, scal, out);
}
